// Merge_Block_56100862820794
// MI455X (gfx1250) — matrix operand order repaired and checked by reading, not compiled, not run on hardware
//
#include <hip/hip_runtime.h>
#include <hip/hip_fp16.h>
#include <math.h>

typedef __attribute__((ext_vector_type(16))) _Float16     v16h;
typedef __attribute__((ext_vector_type(8)))  float        v8f;
typedef __attribute__((ext_vector_type(4)))  unsigned int v4u;
typedef __attribute__((ext_vector_type(8)))  int          v8i;
typedef __attribute__((ext_vector_type(4)))  int          v4i;

#define BATCH   2
#define CIN     55
#define HDIM    768
#define WDIM    768
#define NRSEG   24
#define NCSEG   24
#define CTOPN   36
#define KTOT    1375          /* 55*25 */
#define KCH     43            /* ceil(1375/32) */
#define TILE_H  4
#define TILE_W  64
#define LROWS   16            /* TILE_H + 2*6 */
#define LCOLS   80            /* TILE_W + 12, padded to 80 */
#define XT_ELEMS (CIN*LROWS*LCOLS)   /* 70400 halves */
#define XT_PAD   384                 /* zero region for K padding */
#define TBL_ELEMS (3*1376)

/* packed conv-weight A fragments: [ch][br][lane][16 halves] */
#define AFRAG_HALVES (KCH*3*32*16)          /* 66048 */
#define AFRAG_BYTES  (AFRAG_HALVES*2)       /* 132096 */

#define LDS_AFRAG_OFF 0
#define LDS_TBL_OFF   AFRAG_BYTES                     /* 132096 */
#define LDS_XT_OFF    (AFRAG_BYTES + TBL_ELEMS*4)     /* 148608 */
#define LDS_BYTES     (LDS_XT_OFF + (XT_ELEMS+XT_PAD)*2)  /* 290176 */

#define WS_TOPSUM 0                       /* 2*36*24*24 = 41472 floats */
#define WS_BOTSUM 41472                   /* 576 floats */
#define WS_NZERO  42048
#define WS_AFRAG  42048                   /* float offset; byte 168192, 8B aligned */

#if defined(__has_builtin)
#if __has_builtin(__builtin_amdgcn_tensor_load_to_lds)
#define HAVE_TDM 1
#endif
#endif
#ifndef HAVE_TDM
#define HAVE_TDM 0
#endif

// ---------------------------------------------------------------------------
__global__ void zero_ws_kernel(float* p, int n) {
    int i = blockIdx.x * blockDim.x + threadIdx.x;
    if (i < n) p[i] = 0.f;
}

// ---------------------------------------------------------------------------
// Pack the three 5x5x55 conv weights into WMMA A-fragment order (f16), once.
// Layout matches the conv kernel's per-chunk read: [ch][br][lane][j].
__global__ void pack_weights_kernel(const float* __restrict__ w1,
                                    const float* __restrict__ w2,
                                    const float* __restrict__ w3,
                                    _Float16* __restrict__ afrag) {
    int idx = blockIdx.x * blockDim.x + threadIdx.x;
    if (idx >= AFRAG_HALVES) return;
    int j    = idx & 15;
    int lane = (idx >> 4) & 31;
    int br   = (idx >> 9) % 3;
    int ch   = idx / (3 * 512);
    int n  = lane & 15;
    int hs = lane >> 4;
    int kk = ch * 32 + hs * 16 + j;   // A 16x32 f16 layout (contiguous k = 16h + j)
    const float* w = (br == 0) ? w1 : ((br == 1) ? w2 : w3);
    float v = (n < 6 && kk < KTOT) ? w[n * KTOT + kk] : 0.f;
    afrag[idx] = (_Float16)v;
}

// ---------------------------------------------------------------------------
// Fused: 3 dilated 5x5 convs (WMMA implicit GEMM) + ReLU + 18->37 1x1 (WMMA)
// + sigmoid + per-grid-cell sum accumulation. A-fragments DMA'd via TDM.
__global__ __launch_bounds__(256)
void conv_wmma_kernel(const float* __restrict__ x,
                      const int* __restrict__ row_ids,
                      const int* __restrict__ col_ids,
                      const float* __restrict__ pb1,
                      const float* __restrict__ pb2,
                      const float* __restrict__ pb3,
                      const float* __restrict__ wtop, const float* __restrict__ btop,
                      const float* __restrict__ wbot, const float* __restrict__ bbot,
                      const _Float16* __restrict__ afrag_src,
                      float* __restrict__ out,
                      float* __restrict__ ws_top, float* __restrict__ ws_bot) {
    extern __shared__ char smem[];
    const v16h* afrag = (const v16h*)(smem + LDS_AFRAG_OFF);
    int*        tbl   = (int*)(smem + LDS_TBL_OFF);
    _Float16*   xt    = (_Float16*)(smem + LDS_XT_OFF);

    const int tid = threadIdx.x;
    const int bz  = blockIdx.z;
    const int x0  = blockIdx.x * TILE_W;
    const int y0  = blockIdx.y * TILE_H;
    const int wv  = tid >> 5;

    // ---- stage packed A fragments into LDS via the Tensor Data Mover ----
#if HAVE_TDM
    if (wv == 0) {
        unsigned long long ga = (unsigned long long)(const void*)afrag_src;
        const unsigned n8 = AFRAG_BYTES / 8;          // 16512 8-byte elements
        v4u g0;
        g0[0] = 1u;                                   // count=1, user-mode D#
        g0[1] = (unsigned)LDS_AFRAG_OFF;              // lds_addr (bytes)
        g0[2] = (unsigned)(ga & 0xFFFFFFFFu);         // global_addr[31:0]
        g0[3] = (unsigned)((ga >> 32) & 0x01FFFFFFu)  // global_addr[56:32]
                | 0x80000000u;                        // type=2 ("image")
        v8i g1;
        g1[0] = (int)(3u << 16);                      // wg_mask=0, data_size=8B
        g1[1] = (int)((n8 & 0xFFFFu) << 16);          // tensor_dim0[15:0]
        g1[2] = (int)((n8 >> 16) & 0xFFFFu)           // tensor_dim0[31:16]
                | (int)(1u << 16);                    // tensor_dim1[15:0] = 1
        g1[3] = (int)((n8 & 0xFFFFu) << 16);          // tile_dim0 = n8
        g1[4] = 1;                                    // tile_dim1 = 1
        g1[5] = (int)n8;                              // tensor_dim0_stride
        g1[6] = 0;
        g1[7] = 0;
        v4i gz = {0, 0, 0, 0};
#if __clang_major__ >= 23
        v8i gz8 = {0, 0, 0, 0, 0, 0, 0, 0};
        __builtin_amdgcn_tensor_load_to_lds(g0, g1, gz, gz, gz8, 0);
#else
        __builtin_amdgcn_tensor_load_to_lds(g0, g1, gz, gz, 0);
#endif
#if defined(__has_builtin) && __has_builtin(__builtin_amdgcn_s_wait_tensorcnt)
        __builtin_amdgcn_s_wait_tensorcnt(0);
#else
        asm volatile("s_wait_tensorcnt 0x0" ::: "memory");
#endif
    }
#else
    {
        const unsigned* src = (const unsigned*)(const void*)afrag_src;
        unsigned* dst = (unsigned*)(smem + LDS_AFRAG_OFF);
        for (int i = tid; i < AFRAG_BYTES / 4; i += 256) dst[i] = src[i];
    }
#endif

    // ---- build per-branch K -> LDS-offset LUT (K order: ci*25 + kh*5 + kw) ----
    for (int i = tid; i < TBL_ELEMS; i += 256) {
        int br = i / 1376, k = i - br * 1376;
        int off;
        if (k < KTOT) {
            int ci = k / 25, rem = k - ci * 25;
            int kh = rem / 5, kw = rem - kh * 5;
            int dil = br + 1;
            off = ci * (LROWS * LCOLS) + (6 + dil * (kh - 2)) * LCOLS
                                       + (6 + dil * (kw - 2));
        } else {
            off = XT_ELEMS;                     // zero pad region
        }
        tbl[i] = off;
    }

    // ---- stage x tile to LDS as f16 (zero-padded halo) ----
    const float* xb = x + (size_t)bz * CIN * HDIM * WDIM;
    for (int i = tid; i < XT_ELEMS + XT_PAD; i += 256) {
        float v = 0.f;
        if (i < XT_ELEMS) {
            int ci = i / (LROWS * LCOLS);
            int r  = (i / LCOLS) % LROWS;
            int c  = i % LCOLS;
            int gr = y0 - 6 + r, gc = x0 - 6 + c;
            if (c < TILE_W + 12 && gr >= 0 && gr < HDIM && gc >= 0 && gc < WDIM)
                v = xb[((size_t)ci * HDIM + gr) * WDIM + gc];
        }
        xt[i] = (_Float16)v;
    }
    __syncthreads();

    const int lane = tid & 31;
    const int n    = lane & 15;     // N (pixel) index / M row index for A frags
    const int hs   = lane >> 4;     // half select

    const float* bB[3] = { pb1, pb2, pb3 };

    // ---- A fragments for the fused 37x18 1x1 (rows 0..35 = w_top, 36 = w_bot) ----
    v16h at[3];
#pragma unroll
    for (int mt = 0; mt < 3; ++mt) {
        int row = mt * 16 + n;
#pragma unroll
        for (int j = 0; j < 16; ++j) {
            int kk = j + hs * 16;
            float wvf = 0.f;
            if (kk < 18 && row < 37)
                wvf = (row < 36) ? wtop[row * 18 + kk] : wbot[kk];
            at[mt][j] = (_Float16)wvf;
        }
    }

    for (int t2 = 0; t2 < 2; ++t2) {
        const int tile   = wv * 2 + t2;
        const int py     = tile >> 2;
        const int px     = ((tile & 3) << 4) + n;
        const int pixoff = py * LCOLS + px;

        // ---- implicit-GEMM conv: D[6ch x 16px] over K=1375 per branch ----
        v8f acc[3] = { (v8f){}, (v8f){}, (v8f){} };
        for (int ch = 0; ch < KCH; ++ch) {
            const int kbB = ch * 32 + hs * 16;
#pragma unroll
            for (int br = 0; br < 3; ++br) {
                v16h a = afrag[(ch * 3 + br) * 32 + lane];  // 32B LDS vector load
                v16h bfr;
#pragma unroll
                for (int j = 0; j < 16; ++j)
                    bfr[j] = xt[tbl[br * 1376 + (kbB + j)] + pixoff];
                acc[br] = __builtin_amdgcn_wmma_f32_16x16x32_f16(
                    false, a, false, bfr, (short)0, acc[br], false, false);
            }
        }

        const int h    = y0 + py;
        const int wcol = x0 + px;

        // ---- bias + ReLU, write out_feature (channels 36..53) ----
        float ff[18];
#pragma unroll
        for (int br = 0; br < 3; ++br)
#pragma unroll
            for (int m = 0; m < 6; ++m)
                ff[br * 6 + m] = fmaxf(acc[br][m] + bB[br][m], 0.f);

        if (hs == 0) {
            size_t pbase = (size_t)bz * 55 * HDIM * WDIM + (size_t)h * WDIM + wcol;
#pragma unroll
            for (int k = 0; k < 18; ++k)
                out[pbase + (size_t)(36 + k) * HDIM * WDIM] = ff[k];
        }

        // ---- B fragment of features [18 x 16px] for the 1x1 WMMA ----
        float f16v = __shfl(ff[16], n, 32);
        float f17v = __shfl(ff[17], n, 32);
        v16h bf1;
#pragma unroll
        for (int j = 0; j < 16; ++j) {
            float lo = ff[j];                                     // hs==0: K=j
            float hi = (j == 0) ? f16v : ((j == 1) ? f17v : 0.f); // hs==1: K=16+j
            bf1[j] = (_Float16)(hs ? hi : lo);
        }

        v8f t[3];
#pragma unroll
        for (int mt = 0; mt < 3; ++mt) {
            v8f z = {};
            t[mt] = __builtin_amdgcn_wmma_f32_16x16x32_f16(
                false, at[mt], false, bf1, (short)0, z, false, false);
        }

        // ---- per-grid-cell accumulation ----
        int r = row_ids[h];            // uniform within tile row
        int c = col_ids[wcol];         // per lane (pattern identical in halves)
        int cref = __shfl(c, hs * 16, 32);
        bool uni = __all(c == cref);

#pragma unroll
        for (int mt = 0; mt < 3; ++mt) {
#pragma unroll
            for (int v = 0; v < 8; ++v) {
                int o = mt * 16 + hs * 8 + v;
                if (o < 37) {
                    float val = t[mt][v];
                    if (o < 36) {
                        val += btop[o];
                    } else {
                        val += bbot[0];
                        val = 1.f / (1.f + __expf(-val));   // sigmoid
                    }
                    if (uni) {
                        float s = val;
                        s += __shfl_xor(s, 1, 32);
                        s += __shfl_xor(s, 2, 32);
                        s += __shfl_xor(s, 4, 32);
                        s += __shfl_xor(s, 8, 32);
                        if (n == 0) {
                            if (o < 36)
                                atomicAdd(&ws_top[((bz * CTOPN + o) * NRSEG + r) * NCSEG + c], s);
                            else
                                atomicAdd(&ws_bot[r * NCSEG + c], s);
                        }
                    } else {
                        if (o < 36)
                            atomicAdd(&ws_top[((bz * CTOPN + o) * NRSEG + r) * NCSEG + c], val);
                        else
                            atomicAdd(&ws_bot[r * NCSEG + c], val);
                    }
                }
            }
        }
    }
}

// ---------------------------------------------------------------------------
// Turn sums into means (in place) and emit pools to the output tail.
__global__ void finalize_kernel(const int* __restrict__ row_ids,
                                const int* __restrict__ col_ids,
                                float* __restrict__ ws_top,   // in-place -> means
                                float* __restrict__ ws_bot,   // in-place -> pools
                                float* __restrict__ out) {
    __shared__ int cnts[NRSEG + NCSEG];
    int tid = threadIdx.x;
    if (tid < NRSEG + NCSEG) cnts[tid] = 0;
    __syncthreads();
    for (int i = tid; i < HDIM; i += blockDim.x) atomicAdd(&cnts[row_ids[i]], 1);
    for (int i = tid; i < WDIM; i += blockDim.x) atomicAdd(&cnts[NRSEG + col_ids[i]], 1);
    __syncthreads();

    for (int cell = tid; cell < NRSEG * NCSEG; cell += blockDim.x) {
        int r = cell / NCSEG, c = cell - r * NCSEG;
        float cnt = fmaxf((float)cnts[r] * (float)cnts[NRSEG + c], 1.f);
        float inv = 1.f / cnt;
        for (int bo = 0; bo < BATCH * CTOPN; ++bo)
            ws_top[bo * NRSEG * NCSEG + cell] *= inv;
        float pv = ws_bot[cell] * (inv * (1.f / (float)BATCH));
        ws_bot[cell] = pv;
        size_t base = (size_t)BATCH * 55 * HDIM * WDIM;
        out[base + cell] = pv;                       // pools, b=0
        out[base + NRSEG * NCSEG + cell] = pv;       // pools, b=1
    }
}

// ---------------------------------------------------------------------------
// Broadcast cell means (ch 0..35) and pools (ch 54) back to full resolution.
__global__ void broadcast_kernel(const int* __restrict__ row_ids,
                                 const int* __restrict__ col_ids,
                                 const float* __restrict__ means,
                                 const float* __restrict__ pools,
                                 float* __restrict__ out) {
    int idx = blockIdx.x * blockDim.x + threadIdx.x;      // over B*H*W
    if (idx >= BATCH * HDIM * WDIM) return;
    int w = idx % WDIM;
    int h = (idx / WDIM) % HDIM;
    int b = idx / (HDIM * WDIM);
    int r = row_ids[h], c = col_ids[w];
    int cell = r * NCSEG + c;
    size_t pbase = (size_t)b * 55 * HDIM * WDIM + (size_t)h * WDIM + w;
#pragma unroll
    for (int o = 0; o < CTOPN; ++o)
        out[pbase + (size_t)o * HDIM * WDIM] =
            means[(b * CTOPN + o) * NRSEG * NCSEG + cell];
    out[pbase + (size_t)54 * HDIM * WDIM] = pools[cell];
}

// ---------------------------------------------------------------------------
extern "C" void kernel_launch(void* const* d_in, const int* in_sizes, int n_in,
                              void* d_out, int out_size, void* d_ws, size_t ws_size,
                              hipStream_t stream) {
    const float* x       = (const float*)d_in[0];
    const int*   row_ids = (const int*)d_in[1];
    const int*   col_ids = (const int*)d_in[2];
    const float* w1   = (const float*)d_in[5];
    const float* pb1  = (const float*)d_in[6];
    const float* w2   = (const float*)d_in[7];
    const float* pb2  = (const float*)d_in[8];
    const float* w3   = (const float*)d_in[9];
    const float* pb3  = (const float*)d_in[10];
    const float* wtop = (const float*)d_in[11];
    const float* btop = (const float*)d_in[12];
    const float* wbot = (const float*)d_in[13];
    const float* bbot = (const float*)d_in[14];

    float*    out     = (float*)d_out;
    float*    ws      = (float*)d_ws;
    float*    ws_top  = ws + WS_TOPSUM;
    float*    ws_bot  = ws + WS_BOTSUM;
    _Float16* afrag   = (_Float16*)(ws + WS_AFRAG);

    zero_ws_kernel<<<(WS_NZERO + 255) / 256, 256, 0, stream>>>(ws, WS_NZERO);
    pack_weights_kernel<<<(AFRAG_HALVES + 255) / 256, 256, 0, stream>>>(
        w1, w2, w3, afrag);

    dim3 grid(WDIM / TILE_W, HDIM / TILE_H, BATCH);
    conv_wmma_kernel<<<grid, 256, LDS_BYTES, stream>>>(
        x, row_ids, col_ids, pb1, pb2, pb3,
        wtop, btop, wbot, bbot, afrag, out, ws_top, ws_bot);

    finalize_kernel<<<1, 256, 0, stream>>>(row_ids, col_ids, ws_top, ws_bot, out);

    int npix = BATCH * HDIM * WDIM;
    broadcast_kernel<<<(npix + 255) / 256, 256, 0, stream>>>(
        row_ids, col_ids, ws_top, ws_bot, out);
}
